// UnimodelAttention_50002009260176
// MI455X (gfx1250) — matrix operand order repaired and checked by reading, not compiled, not run on hardware
//
#include <hip/hip_runtime.h>
#include <hip/hip_bf16.h>

#define LL 128
#define BB 2048
#define DD 256
#define TT 64
#define VV 128
#define VP (VV + 8)   // padded sXT row stride in halfs (272 B, 16B-aligned)

typedef __attribute__((ext_vector_type(16))) __bf16 v16bf;
typedef __attribute__((ext_vector_type(8)))  float  v8f;

__device__ __forceinline__ float sigmoidf(float x) {
    return 1.0f / (1.0f + __expf(-x));
}

// ---------------------------------------------------------------------------
// Kernel 1: one block per batch b.
//   resized = rownorm(assign) @ seq_sel      (bf16 WMMA, K=128)
//   h_att   = sigmoid(mean_d(resized) @ W_h^T + b_h)
//   w_att   = sigmoid(mean_t(resized) @ W_w^T + b_w)
//   h_rec   = colnorm(assign)^T @ h_att
// writes h_rec [B,128] and w_att [B,256] to workspace.
// ---------------------------------------------------------------------------
__global__ __launch_bounds__(256)
void uma_gates_kernel(const float* __restrict__ x,
                      const int*   __restrict__ v_len,
                      const float* __restrict__ assign,
                      const float* __restrict__ W_h, const float* __restrict__ b_h,
                      const float* __restrict__ W_w, const float* __restrict__ b_w,
                      float* __restrict__ ws_hrec,   // [B,128]
                      float* __restrict__ ws_watt) { // [B,256]
    __shared__ __align__(16) __bf16 sXT[DD][VP];   // seq_sel^T bf16    ~68 KB
    __shared__ __align__(16) float  sRes[TT][DD];  // resized (+staging) 64 KB
    __shared__ __align__(16) __bf16 sAbf[TT][VV];  // rownorm(A) bf16    16 KB
    __shared__ float sInvRow[TT];
    __shared__ float sColSum[VV];
    __shared__ float sRowMean[TT];
    __shared__ float sColMean[DD];
    __shared__ float sHatt[TT];

    const int tid = threadIdx.x;
    const int b   = blockIdx.x;
    const int m   = v_len[b];
    const float* Ab = assign + (size_t)b * (TT * VV);

    // ---- stage raw assign into sRes scratch (8192 floats) ----
    float* stage = &sRes[0][0];
    #pragma unroll
    for (int k = 0; k < 32; ++k) stage[k * 256 + tid] = Ab[k * 256 + tid];

    // ---- load seq_sel TRANSPOSED: thread owns column d = tid ----
    // x[v, b, d] for v < m (else 0); 8 v's packed per ds_store_b128.
    {
        const float* xcol = x + (size_t)b * DD + tid;  // + v * (BB*DD)
        #pragma unroll
        for (int vg = 0; vg < 16; ++vg) {
            union { __bf16 h[8]; uint4 u; } pk;
            #pragma unroll
            for (int j = 0; j < 8; ++j) {
                const int v = vg * 8 + j;
                const float val = xcol[(size_t)v * (BB * DD)];
                pk.h[j] = (v < m) ? (__bf16)val : (__bf16)0.f;
            }
            *(uint4*)&sXT[tid][vg * 8] = pk.u;
        }
    }
    __syncthreads();

    // ---- row sums (over v) and column sums (over t) of assign ----
    if (tid < TT) {
        float s = 0.f;
        #pragma unroll 8
        for (int v = 0; v < VV; ++v) s += stage[tid * VV + v];
        sInvRow[tid] = 1.0f / fmaxf(s, 1.0f);
    }
    if (tid < VV) {
        float s = 0.f;
        #pragma unroll 8
        for (int t = 0; t < TT; ++t) s += stage[t * VV + tid];
        sColSum[tid] = s;
    }
    __syncthreads();

    // ---- build row-normalized A in bf16 ----
    #pragma unroll
    for (int k = 0; k < 32; ++k) {
        int idx = k * 256 + tid;
        int t = idx >> 7;
        (&sAbf[0][0])[idx] = (__bf16)(stage[idx] * sInvRow[t]);
    }
    __syncthreads();

    // ---- WMMA matmul: resized[64][256] = A_rs[64][128] @ X[128][256] ----
    {
        const int wave = tid >> 5;
        const int lane = tid & 31;
        const int half = lane >> 4;       // 0/1
        const int mrow = lane & 15;
        const int n    = lane & 15;
        const int m4   = wave & 3;        // M tile (rows 16*m4..)
        const int nb   = (wave >> 2) * 8; // 8 N tiles per wave

        // A fragments: two contiguous 8-half runs per K-step (b128 loads)
        v16bf afrag[4];
        const __bf16* arow = &sAbf[16 * m4 + mrow][0];
        #pragma unroll
        for (int kk = 0; kk < 4; ++kk) {
            union { uint4 u[2]; v16bf v; } au;
            au.u[0] = *(const uint4*)(arow + kk * 32 + half * 8);
            au.u[1] = *(const uint4*)(arow + kk * 32 + 16 + half * 8);
            afrag[kk] = au.v;
        }
        #pragma unroll
        for (int nt = 0; nt < 8; ++nt) {
            const int ncol = (nb + nt) * 16 + n;
            const __bf16* brow = &sXT[ncol][0];
            v8f c = {};
            #pragma unroll
            for (int kk = 0; kk < 4; ++kk) {
                // B fragment: two contiguous 8-half runs of the transposed row
                union { uint4 u[2]; v16bf v; } bu;
                bu.u[0] = *(const uint4*)(brow + kk * 32 + half * 8);
                bu.u[1] = *(const uint4*)(brow + kk * 32 + 16 + half * 8);
                c = __builtin_amdgcn_wmma_f32_16x16x32_bf16(
                        false, afrag[kk], false, bu.v, (short)0, c, false, false);
            }
            #pragma unroll
            for (int r = 0; r < 8; ++r)
                sRes[16 * m4 + half * 8 + r][ncol] = c[r];
        }
    }
    __syncthreads();

    // ---- means over d (rows) and over t (cols) ----
    if (tid < TT) {
        float s = 0.f;
        #pragma unroll 8
        for (int d = 0; d < DD; ++d) s += sRes[tid][d];
        sRowMean[tid] = s * (1.0f / DD);
    }
    {
        float s = 0.f;
        #pragma unroll 8
        for (int t = 0; t < TT; ++t) s += sRes[t][tid];
        sColMean[tid] = s * (1.0f / TT);
    }
    __syncthreads();

    // ---- gates ----
    if (tid < TT) {
        float acc = b_h[tid];
        #pragma unroll 8
        for (int j = 0; j < TT; ++j) acc += sRowMean[j] * W_h[tid * TT + j];
        sHatt[tid] = sigmoidf(acc);
    }
    {
        float acc = b_w[tid];
        #pragma unroll 8
        for (int j = 0; j < DD; ++j) acc += sColMean[j] * W_w[tid * DD + j];
        ws_watt[(size_t)b * DD + tid] = sigmoidf(acc);
    }
    __syncthreads();

    // ---- h_rec: column-normalized assign^T @ h_att (re-read assign) ----
    if (tid < VV) {
        float acc = 0.f;
        #pragma unroll 8
        for (int t = 0; t < TT; ++t) acc += Ab[t * VV + tid] * sHatt[t];
        ws_hrec[(size_t)b * VV + tid] = acc / fmaxf(sColSum[tid], 1.0f);
    }
}

// ---------------------------------------------------------------------------
// Kernel 2: streaming elementwise pass over x [L,B,D].
// Each block = 4 (l,b) rows; each thread a float4 of the D dimension.
// ---------------------------------------------------------------------------
__global__ __launch_bounds__(256)
void uma_apply_kernel(const float* __restrict__ x,
                      const int*   __restrict__ v_len,
                      const float* __restrict__ ws_hrec,
                      const float* __restrict__ ws_watt,
                      float* __restrict__ out) {
    const int tid = threadIdx.x;
    const int row = blockIdx.x * 4 + (tid >> 6);   // flat l*B + b
    const int l   = row >> 11;                     // B = 2048
    const int b   = row & (BB - 1);
    const int d   = (tid & 63) * 4;
    const size_t off = (size_t)row * DD + d;

    float4 xv = *(const float4*)(x + off);
    if (l < v_len[b]) {
        const float hr = ws_hrec[(size_t)b * VV + l];
        const float4 wv = *(const float4*)(ws_watt + (size_t)b * DD + d);
        xv.x *= 0.5f * (hr + wv.x);
        xv.y *= 0.5f * (hr + wv.y);
        xv.z *= 0.5f * (hr + wv.z);
        xv.w *= 0.5f * (hr + wv.w);
    }
    *(float4*)(out + off) = xv;
}

// ---------------------------------------------------------------------------
// Kernel 3: pass is_rnn through (as float, appended after x_new).
// ---------------------------------------------------------------------------
__global__ __launch_bounds__(256)
void uma_isrnn_kernel(const int* __restrict__ is_rnn, float* __restrict__ out) {
    const int i = blockIdx.x * blockDim.x + threadIdx.x;
    out[i] = (float)is_rnn[i];
}

extern "C" void kernel_launch(void* const* d_in, const int* in_sizes, int n_in,
                              void* d_out, int out_size, void* d_ws, size_t ws_size,
                              hipStream_t stream) {
    const float* x      = (const float*)d_in[0];
    const int*   is_rnn = (const int*)  d_in[1];
    const int*   v_len  = (const int*)  d_in[2];
    // d_in[3] = sel_idx (arange layout; implied by v_len)
    const float* assign = (const float*)d_in[4];
    const float* W_h    = (const float*)d_in[5];
    const float* b_h    = (const float*)d_in[6];
    const float* W_w    = (const float*)d_in[7];
    const float* b_w    = (const float*)d_in[8];

    float* ws_hrec = (float*)d_ws;                    // B*128 floats
    float* ws_watt = ws_hrec + (size_t)BB * VV;       // B*256 floats

    float* out_x  = (float*)d_out;                    // L*B*D
    float* out_ir = out_x + (size_t)LL * BB * DD;     // L*B

    uma_gates_kernel<<<BB, 256, 0, stream>>>(x, v_len, assign, W_h, b_h, W_w, b_w,
                                             ws_hrec, ws_watt);
    uma_apply_kernel<<<(LL * BB) / 4, 256, 0, stream>>>(x, v_len, ws_hrec, ws_watt, out_x);
    uma_isrnn_kernel<<<(LL * BB) / 256, 256, 0, stream>>>(is_rnn, out_ir);
}
